// BitLinearx_24962349924855
// MI455X (gfx1250) — 8-bit matrix operand order repaired and checked by reading, compiled, not run on hardware
//
#include <hip/hip_runtime.h>

typedef int v8i __attribute__((ext_vector_type(8)));

#define T_TOKENS 8192
#define D_IN     4096
#define D_OUT    11008
#define W_ELEMS  (D_OUT * D_IN)
#define X_ELEMS  (T_TOKENS * D_IN)
#define NSTEP    (D_IN / 64)

#define WS_SRED_OFF 0
#define WS_RS_OFF   256
#define WS_QA_OFF   (256 + 32768)
#define WS_TW_OFF   (WS_QA_OFF + X_ELEMS)

#if defined(__has_builtin)
#if __has_builtin(__builtin_amdgcn_global_load_async_to_lds_b128) && \
    __has_builtin(__builtin_amdgcn_s_wait_asynccnt)
#define USE_ASYNC_LDS 1
#endif
#endif
#ifndef USE_ASYNC_LDS
#define USE_ASYNC_LDS 0
#endif

typedef int v4b __attribute__((vector_size(16)));
typedef __attribute__((address_space(1))) v4b* g_v4b_p;
typedef __attribute__((address_space(3))) v4b* l_v4b_p;
#define ASYNC_CP16(gsrc, ldst) \
    __builtin_amdgcn_global_load_async_to_lds_b128((g_v4b_p)(gsrc), (l_v4b_p)(ldst), 0, 0)

__global__ void bitlinear_init(float* __restrict__ sred) {
    sred[0] = 0.0f;
}

__global__ __launch_bounds__(256)
void bitlinear_wabs(const float* __restrict__ w, float* __restrict__ sred) {
    __shared__ float red[256];
    const int tid = threadIdx.x;
    const float4* w4 = (const float4*)w;
    const int n4 = W_ELEMS / 4;
    float s = 0.0f;
    for (int i = blockIdx.x * 256 + tid; i < n4; i += gridDim.x * 256) {
        const float4 v = w4[i];
        s += fabsf(v.x) + fabsf(v.y) + fabsf(v.z) + fabsf(v.w);
    }
    red[tid] = s;
    __syncthreads();
    for (int st = 128; st > 0; st >>= 1) {
        if (tid < st) red[tid] += red[tid + st];
        __syncthreads();
    }
    if (tid == 0) atomicAdd(&sred[0], red[0]);
}

__global__ void bitlinear_finalize(float* __restrict__ sred) {
    const float mean = sred[0] / (float)W_ELEMS;
    sred[1] = 1.0f / fmaxf(mean, 1e-5f);
}

__global__ __launch_bounds__(256)
void bitlinear_quant_w(const float* __restrict__ w,
                       signed char* __restrict__ tw,
                       const float* __restrict__ sred) {
    const float sw = sred[1];
    const int i = blockIdx.x * 256 + threadIdx.x;
    const float4 v = ((const float4*)w)[i];
    char4 c;
    c.x = (signed char)fminf(fmaxf(rintf(v.x * sw), -1.0f), 1.0f);
    c.y = (signed char)fminf(fmaxf(rintf(v.y * sw), -1.0f), 1.0f);
    c.z = (signed char)fminf(fmaxf(rintf(v.z * sw), -1.0f), 1.0f);
    c.w = (signed char)fminf(fmaxf(rintf(v.w * sw), -1.0f), 1.0f);
    ((char4*)tw)[i] = c;
}

__global__ __launch_bounds__(256)
void bitlinear_quant_act(const float* __restrict__ x,
                         signed char* __restrict__ qA,
                         float* __restrict__ rowScale) {
    __shared__ float red[256];
    const int row = blockIdx.x;
    const int tid = threadIdx.x;
    const float4* xr = (const float4*)(x + (size_t)row * D_IN);
    float4 v[4];
    float m = 0.0f;
#pragma unroll
    for (int j = 0; j < 4; ++j) {
        v[j] = xr[tid + j * 256];
        m = fmaxf(m, fmaxf(fmaxf(fabsf(v[j].x), fabsf(v[j].y)),
                           fmaxf(fabsf(v[j].z), fabsf(v[j].w))));
    }
    red[tid] = m;
    __syncthreads();
    for (int st = 128; st > 0; st >>= 1) {
        if (tid < st) red[tid] = fmaxf(red[tid], red[tid + st]);
        __syncthreads();
    }
    const float amax = fmaxf(red[0], 1e-5f);
    const float sact = 127.0f / amax;
    if (tid == 0) rowScale[row] = (amax + 2e-6f) / 127.0f;
    char4* qr = (char4*)(qA + (size_t)row * D_IN);
#pragma unroll
    for (int j = 0; j < 4; ++j) {
        char4 c;
        c.x = (signed char)fminf(fmaxf(rintf(v[j].x * sact), -128.0f), 127.0f);
        c.y = (signed char)fminf(fmaxf(rintf(v[j].y * sact), -128.0f), 127.0f);
        c.z = (signed char)fminf(fmaxf(rintf(v[j].z * sact), -128.0f), 127.0f);
        c.w = (signed char)fminf(fmaxf(rintf(v[j].w * sact), -128.0f), 127.0f);
        qr[tid + j * 256] = c;
    }
}

__global__ __launch_bounds__(256)
void bitlinear_gemm_wmma(const signed char* __restrict__ qA,
                         const signed char* __restrict__ tW,
                         const float* __restrict__ rowScale,
                         const float* __restrict__ sred,
                         float* __restrict__ out) {
    __shared__ __align__(16) signed char As[2][128][64];
    __shared__ __align__(16) signed char Bs[2][128][64];

    const int tid  = threadIdx.x;
    const int lane = tid & 31;
    const int wv   = tid >> 5;
    const int wm   = wv & 3;
    const int wn   = wv >> 2;
    const int blockM = blockIdx.y * 128;
    const int blockN = blockIdx.x * 128;
    const float sw = sred[1];

    const int row0 = tid >> 2,         koff0 = (tid & 3) << 4;
    const int row1 = (tid + 256) >> 2, koff1 = (tid & 3) << 4;
    const signed char* gA0 = qA + (size_t)(blockM + row0) * D_IN + koff0;
    const signed char* gA1 = qA + (size_t)(blockM + row1) * D_IN + koff1;
    const signed char* gB0 = tW + (size_t)(blockN + row0) * D_IN + koff0;
    const signed char* gB1 = tW + (size_t)(blockN + row1) * D_IN + koff1;

    const v8i vzero = {0, 0, 0, 0, 0, 0, 0, 0};
    v8i acc[2][4];
#pragma unroll
    for (int mi = 0; mi < 2; ++mi)
#pragma unroll
        for (int ni = 0; ni < 4; ++ni)
            acc[mi][ni] = vzero;

    const int laneM = lane & 15;
    const int kbA   = (lane >> 4) * 16;
    const int kbB   = (lane >> 4) * 16;

    auto compute = [&](int buf) {
        v8i afrag[2];
#pragma unroll
        for (int mi = 0; mi < 2; ++mi) {
            const int r = wm * 32 + mi * 16 + laneM;
            const int4 p0 = *(const int4*)(&As[buf][r][kbA]);
            const int4 p1 = *(const int4*)(&As[buf][r][kbA + 32]);
            v8i a = {p0.x, p0.y, p0.z, p0.w, p1.x, p1.y, p1.z, p1.w};
            afrag[mi] = a;
        }
#pragma unroll
        for (int ni = 0; ni < 4; ++ni) {
            const int n = wn * 64 + ni * 16 + laneM;
            const int4 q0 = *(const int4*)(&Bs[buf][n][kbB]);
            const int4 q1 = *(const int4*)(&Bs[buf][n][kbB + 32]);
            v8i b = {q0.x, q0.y, q0.z, q0.w, q1.x, q1.y, q1.z, q1.w};
#pragma unroll
            for (int mi = 0; mi < 2; ++mi) {
                acc[mi][ni] = __builtin_amdgcn_wmma_i32_16x16x64_iu8(
                     true, afrag[mi],
                     true, b,
                    acc[mi][ni],  false,  false);
            }
        }
    };

#if USE_ASYNC_LDS
    ASYNC_CP16(gA0, &As[0][row0][koff0]);
    ASYNC_CP16(gA1, &As[0][row1][koff1]);
    ASYNC_CP16(gB0, &Bs[0][row0][koff0]);
    ASYNC_CP16(gB1, &Bs[0][row1][koff1]);
    __builtin_amdgcn_s_wait_asynccnt(0);
    __syncthreads();

#pragma unroll 1
    for (int step = 0; step < NSTEP - 1; ++step) {
        const int buf  = step & 1;
        const int knxt = (step + 1) * 64;
        ASYNC_CP16(gA0 + knxt, &As[buf ^ 1][row0][koff0]);
        ASYNC_CP16(gA1 + knxt, &As[buf ^ 1][row1][koff1]);
        ASYNC_CP16(gB0 + knxt, &Bs[buf ^ 1][row0][koff0]);
        ASYNC_CP16(gB1 + knxt, &Bs[buf ^ 1][row1][koff1]);

        compute(buf);

        __builtin_amdgcn_s_wait_asynccnt(0);
        __syncthreads();
    }
    compute((NSTEP - 1) & 1);
#else
    *(int4*)(&As[0][row0][koff0]) = *(const int4*)(gA0);
    *(int4*)(&As[0][row1][koff1]) = *(const int4*)(gA1);
    *(int4*)(&Bs[0][row0][koff0]) = *(const int4*)(gB0);
    *(int4*)(&Bs[0][row1][koff1]) = *(const int4*)(gB1);
    __syncthreads();

#pragma unroll 1
    for (int step = 0; step < NSTEP - 1; ++step) {
        const int buf  = step & 1;
        const int knxt = (step + 1) * 64;
        const int4 na0 = *(const int4*)(gA0 + knxt);
        const int4 na1 = *(const int4*)(gA1 + knxt);
        const int4 nb0 = *(const int4*)(gB0 + knxt);
        const int4 nb1 = *(const int4*)(gB1 + knxt);
        __builtin_prefetch(gA0 + knxt + 64, 0, 1);
        __builtin_prefetch(gB0 + knxt + 64, 0, 1);

        compute(buf);

        *(int4*)(&As[buf ^ 1][row0][koff0]) = na0;
        *(int4*)(&As[buf ^ 1][row1][koff1]) = na1;
        *(int4*)(&Bs[buf ^ 1][row0][koff0]) = nb0;
        *(int4*)(&Bs[buf ^ 1][row1][koff1]) = nb1;
        __syncthreads();
    }
    compute((NSTEP - 1) & 1);
#endif

    const int halfAdd = (lane >> 4) * 8;
#pragma unroll
    for (int mi = 0; mi < 2; ++mi) {
#pragma unroll
        for (int ni = 0; ni < 4; ++ni) {
            const int col = blockN + wn * 64 + ni * 16 + laneM;
#pragma unroll
            for (int r = 0; r < 8; ++r) {
                const int row = blockM + wm * 32 + mi * 16 + r + halfAdd;
                const float scale = sw * rowScale[row];
                out[(size_t)row * D_OUT + col] = (float)acc[mi][ni][r] * scale;
            }
        }
    }
}

extern "C" void kernel_launch(void* const* d_in, const int* in_sizes, int n_in,
                              void* d_out, int out_size, void* d_ws, size_t ws_size,
                              hipStream_t stream) {
    const float* x = (const float*)d_in[0];
    const float* w = (const float*)d_in[1];
    float* out = (float*)d_out;

    char* ws = (char*)d_ws;
    float*       sred = (float*)(ws + WS_SRED_OFF);
    float*       rs   = (float*)(ws + WS_RS_OFF);
    signed char* qA   = (signed char*)(ws + WS_QA_OFF);
    signed char* tw   = (signed char*)(ws + WS_TW_OFF);

    bitlinear_init<<<1, 1, 0, stream>>>(sred);
    bitlinear_wabs<<<2048, 256, 0, stream>>>(w, sred);
    bitlinear_finalize<<<1, 1, 0, stream>>>(sred);
    bitlinear_quant_w<<<W_ELEMS / (4 * 256), 256, 0, stream>>>(w, tw, sred);
    bitlinear_quant_act<<<T_TOKENS, 256, 0, stream>>>(x, qA, rs);

    dim3 grid(D_OUT / 128, T_TOKENS / 128);
    bitlinear_gemm_wmma<<<grid, 256, 0, stream>>>(qA, tw, rs, sred, out);
}
